// GCN_63136019251380
// MI455X (gfx1250) — matrix operand order repaired and checked by reading, not compiled, not run on hardware
//
#include <hip/hip_runtime.h>

// ---------------------------------------------------------------------------
// GCN forward for MI455X (gfx1250, wave32):
//   h1 = x @ W1            (bf16 WMMA, fp32 accum)         [N,256]
//   agg1 = A @ h1          (edge-parallel hw f32 atomics,  [N,256]
//                           h1 resident in 192MB L2)
//   h2 = relu(agg1) @ W2   (bf16 WMMA, N padded 40->48)    [N,40]
//   out = A @ h2           (edge-parallel hw f32 atomics)  [N,40]
//   out = log_softmax(out) (wave-per-node, shfl reduce)
// ---------------------------------------------------------------------------

typedef __attribute__((ext_vector_type(16))) __bf16 v16bf;
typedef __attribute__((ext_vector_type(8)))  float  v8f;
typedef __attribute__((ext_vector_type(4)))  float  v4f;

#define NFEAT 512
#define NHID  256
#define NCLS  40

// --- repack W1 (512x256 f32, row-major) into bf16 B-fragments --------------
// Fragment (kt,nt) covers K=[kt*32,kt*32+32), N=[nt*16,nt*16+16).
// Within a fragment: element index = lane*16 + j, value = W1[k][n] with
//   k = kt*32 + (lane/16)*8 + (j&7) + (j>>3)*16,  n = nt*16 + lane%16     (ISA B layout)
__global__ __launch_bounds__(256) void repack_w1(const float* __restrict__ W1,
                                                 __bf16* __restrict__ W1bf) {
  int tid = blockIdx.x * 256 + threadIdx.x;          // 16*16*512 = 131072
  int frag = tid >> 9;
  int r    = tid & 511;
  int L = r >> 4, j = r & 15;
  int kt = frag >> 4, nt = frag & 15;
  int k = kt * 32 + (L >> 4) * 8 + (j & 7) + (j >> 3) * 16;
  int n = nt * 16 + (L & 15);
  W1bf[tid] = (__bf16)W1[k * NHID + n];
}

// --- repack W2 (256x40 f32) into bf16 B-fragments, zero-padded N->48 -------
__global__ __launch_bounds__(256) void repack_w2(const float* __restrict__ W2,
                                                 __bf16* __restrict__ W2bf) {
  int tid = blockIdx.x * 256 + threadIdx.x;          // 8*3*512 = 12288
  int frag = tid >> 9;
  int r    = tid & 511;
  int L = r >> 4, j = r & 15;
  int kt = frag / 3, nt = frag % 3;
  int k = kt * 32 + (L >> 4) * 8 + (j & 7) + (j >> 3) * 16;
  int n = nt * 16 + (L & 15);
  float v = (n < NCLS) ? W2[k * NCLS + n] : 0.0f;
  W2bf[tid] = (__bf16)v;
}

// --- GEMM1: h1[N,256] = x[N,512] @ W1 --------------------------------------
// 4 waves/block; wave w computes the 16(M) x 64(N) tile at columns w*64.
__global__ __launch_bounds__(128) void gemm1(const float* __restrict__ x,
                                             const __bf16* __restrict__ W1bf,
                                             float* __restrict__ h1) {
  const int lane  = threadIdx.x & 31;
  const int wave  = threadIdx.x >> 5;          // 0..3
  const int mbase = blockIdx.x * 16;
  const int m     = mbase + (lane & 15);
  const int sel   = lane >> 4;                 // K half-select (A layout)
  const int nt0   = wave * 4;                  // first of 4 B column tiles

  v8f acc0 = {}, acc1 = {}, acc2 = {}, acc3 = {};

  for (int kt = 0; kt < NFEAT / 32; ++kt) {
    const int kb = kt * 32;
    // A fragment: lane holds row m, K = {kb+8*sel..+7} U {kb+16+8*sel..+7}
    const float* xa = x + (size_t)m * NFEAT + kb + sel * 8;
    v4f f0 = *(const v4f*)(xa);
    v4f f1 = *(const v4f*)(xa + 4);
    v4f f2 = *(const v4f*)(xa + 16);
    v4f f3 = *(const v4f*)(xa + 20);
    v16bf a;
#pragma unroll
    for (int i = 0; i < 4; ++i) {
      a[i]      = (__bf16)f0[i];
      a[4 + i]  = (__bf16)f1[i];
      a[8 + i]  = (__bf16)f2[i];
      a[12 + i] = (__bf16)f3[i];
    }
    // B fragments: contiguous 32B per lane from repacked W1
    const __bf16* wb = W1bf + ((size_t)(kt * 16 + nt0)) * 512 + lane * 16;
    v16bf b0 = *(const v16bf*)(wb);
    v16bf b1 = *(const v16bf*)(wb + 512);
    v16bf b2 = *(const v16bf*)(wb + 1024);
    v16bf b3 = *(const v16bf*)(wb + 1536);

    acc0 = __builtin_amdgcn_wmma_f32_16x16x32_bf16(false, a, false, b0, (short)0, acc0, false, false);
    acc1 = __builtin_amdgcn_wmma_f32_16x16x32_bf16(false, a, false, b1, (short)0, acc1, false, false);
    acc2 = __builtin_amdgcn_wmma_f32_16x16x32_bf16(false, a, false, b2, (short)0, acc2, false, false);
    acc3 = __builtin_amdgcn_wmma_f32_16x16x32_bf16(false, a, false, b3, (short)0, acc3, false, false);
  }

  // C/D layout: M = r + 8*(lane/16), N = lane%16
  const int mrow = mbase + sel * 8;
  const int n0   = wave * 64 + (lane & 15);
#pragma unroll
  for (int r = 0; r < 8; ++r) {
    size_t row = (size_t)(mrow + r) * NHID;
    h1[row + n0]      = acc0[r];
    h1[row + n0 + 16] = acc1[r];
    h1[row + n0 + 32] = acc2[r];
    h1[row + n0 + 48] = acc3[r];
  }
}

// --- GEMM2: h2[N,40] = relu(agg1)[N,256] @ W2 ------------------------------
__global__ __launch_bounds__(128) void gemm2(const float* __restrict__ agg1,
                                             const __bf16* __restrict__ W2bf,
                                             float* __restrict__ h2,
                                             int mtiles) {
  const int lane = threadIdx.x & 31;
  const int wave = threadIdx.x >> 5;
  const int mt   = blockIdx.x * 4 + wave;
  if (mt >= mtiles) return;
  const int mbase = mt * 16;
  const int m     = mbase + (lane & 15);
  const int sel   = lane >> 4;

  v8f acc0 = {}, acc1 = {}, acc2 = {};

  for (int kt = 0; kt < NHID / 32; ++kt) {
    const int kb = kt * 32;
    const float* xa = agg1 + (size_t)m * NHID + kb + sel * 8;
    v4f f0 = *(const v4f*)(xa);
    v4f f1 = *(const v4f*)(xa + 4);
    v4f f2 = *(const v4f*)(xa + 16);
    v4f f3 = *(const v4f*)(xa + 20);
    v16bf a;
#pragma unroll
    for (int i = 0; i < 4; ++i) {              // fused ReLU
      a[i]      = (__bf16)fmaxf(f0[i], 0.0f);
      a[4 + i]  = (__bf16)fmaxf(f1[i], 0.0f);
      a[8 + i]  = (__bf16)fmaxf(f2[i], 0.0f);
      a[12 + i] = (__bf16)fmaxf(f3[i], 0.0f);
    }
    const __bf16* wb = W2bf + ((size_t)(kt * 3)) * 512 + lane * 16;
    v16bf b0 = *(const v16bf*)(wb);
    v16bf b1 = *(const v16bf*)(wb + 512);
    v16bf b2 = *(const v16bf*)(wb + 1024);

    acc0 = __builtin_amdgcn_wmma_f32_16x16x32_bf16(false, a, false, b0, (short)0, acc0, false, false);
    acc1 = __builtin_amdgcn_wmma_f32_16x16x32_bf16(false, a, false, b1, (short)0, acc1, false, false);
    acc2 = __builtin_amdgcn_wmma_f32_16x16x32_bf16(false, a, false, b2, (short)0, acc2, false, false);
  }

  const int mrow = mbase + sel * 8;
  const int n    = lane & 15;
#pragma unroll
  for (int r = 0; r < 8; ++r) {
    size_t row = (size_t)(mrow + r) * NCLS;
    h2[row + n]      = acc0[r];
    h2[row + n + 16] = acc1[r];
    if (n < 8) h2[row + n + 32] = acc2[r];     // N padded 40->48; drop pad
  }
}

// --- SPMM: out[dst] += w * h[src]; one wave per edge -----------------------
// float4 gathers (global_load_b128) + hardware global_atomic_add_f32.
__device__ __forceinline__ void atomic_fadd(float* p, float v) {
  __hip_atomic_fetch_add(p, v, __ATOMIC_RELAXED, __HIP_MEMORY_SCOPE_AGENT);
}

template <int FEAT>
__global__ __launch_bounds__(256) void spmm(const int* __restrict__ src,
                                            const int* __restrict__ dst,
                                            const float* __restrict__ w,
                                            const float* __restrict__ h,
                                            float* __restrict__ out,
                                            int nedges) {
  const int e = blockIdx.x * 8 + (threadIdx.x >> 5);
  if (e >= nedges) return;
  const int lane = threadIdx.x & 31;
  const int s = src[e];
  const int d = dst[e];
  const float ww = w[e];
  const v4f* hs = (const v4f*)(h + (size_t)s * FEAT);
  float* od = out + (size_t)d * FEAT;
  constexpr int NV = FEAT / 4;                 // float4 chunks per row
#pragma unroll
  for (int q0 = 0; q0 < NV; q0 += 32) {
    const int q = q0 + lane;
    if (NV % 32 == 0 || q < NV) {
      v4f v = hs[q];
#pragma unroll
      for (int c = 0; c < 4; ++c)
        atomic_fadd(&od[q * 4 + c], ww * v[c]);
    }
  }
}

// --- log_softmax over 40 classes, one wave per node, in place --------------
__global__ __launch_bounds__(256) void logsm(float* __restrict__ out, int nnodes) {
  const int node = blockIdx.x * 8 + (threadIdx.x >> 5);
  if (node >= nnodes) return;
  const int lane = threadIdx.x & 31;
  float* row = out + (size_t)node * NCLS;
  float v0 = row[lane];
  float v1 = (lane < 8) ? row[32 + lane] : -3.402823466e38f;
  float mx = fmaxf(v0, v1);
#pragma unroll
  for (int off = 16; off > 0; off >>= 1) mx = fmaxf(mx, __shfl_xor(mx, off, 32));
  float e = expf(v0 - mx) + ((lane < 8) ? expf(v1 - mx) : 0.0f);
#pragma unroll
  for (int off = 16; off > 0; off >>= 1) e += __shfl_xor(e, off, 32);
  float ls = mx + logf(e);
  row[lane] = v0 - ls;
  if (lane < 8) row[32 + lane] = v1 - ls;
}

extern "C" void kernel_launch(void* const* d_in, const int* in_sizes, int n_in,
                              void* d_out, int out_size, void* d_ws, size_t ws_size,
                              hipStream_t stream) {
  const float* x    = (const float*)d_in[0];
  const int*   esrc = (const int*)d_in[1];
  const int*   edst = (const int*)d_in[2];
  const float* ew   = (const float*)d_in[3];
  const float* W1   = (const float*)d_in[4];
  const float* W2   = (const float*)d_in[5];
  const int E = in_sizes[1];
  const int N = in_sizes[0] / NFEAT;           // 100000
  float* out = (float*)d_out;

  // workspace: h1/h2 | agg1 | W1bf | W2bf
  char* ws = (char*)d_ws;
  const size_t h1Bytes = (size_t)N * NHID * sizeof(float);
  float*  h1   = (float*)ws;                   // reused as h2 (stride NCLS)
  float*  agg1 = (float*)(ws + h1Bytes);
  __bf16* W1bf = (__bf16*)(ws + 2 * h1Bytes);
  __bf16* W2bf = W1bf + NFEAT * NHID;

  hipMemsetAsync(agg1, 0, h1Bytes, stream);
  hipMemsetAsync(out, 0, (size_t)N * NCLS * sizeof(float), stream);

  repack_w1<<<(NFEAT * NHID) / 256, 256, 0, stream>>>(W1, W1bf);
  repack_w2<<<(8 * 3 * 512) / 256, 256, 0, stream>>>(W2, W2bf);

  const int mtiles = N / 16;                   // 6250
  gemm1<<<mtiles, 128, 0, stream>>>(x, W1bf, h1);
  spmm<NHID><<<(E + 7) / 8, 256, 0, stream>>>(esrc, edst, ew, h1, agg1, E);
  gemm2<<<(mtiles + 3) / 4, 128, 0, stream>>>(agg1, W2bf, h1, mtiles);
  spmm<NCLS><<<(E + 7) / 8, 256, 0, stream>>>(esrc, edst, ew, h1, out, E);
  logsm<<<(N + 7) / 8, 256, 0, stream>>>(out, N);
}
